// FiLM3DECGATBlock_19181323943999
// MI455X (gfx1250) — matrix operand order repaired and checked by reading, not compiled, not run on hardware
//
#include <hip/hip_runtime.h>
#include <hip/hip_bf16.h>
#include <cstdint>

#define B_   4
#define N_   10000
#define E_   160000
#define D_   128
#define H_   4
#define HID_ 128
#define DE_  16
#define DK_  32

typedef __attribute__((ext_vector_type(16))) __bf16 v16bf;
typedef __attribute__((ext_vector_type(8)))  float  v8f;
typedef __attribute__((ext_vector_type(4)))  unsigned v4u;
typedef __attribute__((ext_vector_type(8)))  int    v8i_t;
typedef __attribute__((ext_vector_type(4)))  int    v4i_t;

#if defined(__has_builtin)
#  if __has_builtin(__builtin_amdgcn_tensor_load_to_lds)
#    define HAVE_TDM 1
#  endif
#endif
#ifndef HAVE_TDM
#  define HAVE_TDM 0
#endif

union Frag16 { uint4 q[2]; v16bf v; };

__device__ __forceinline__ unsigned short f2bf(float f) {
  unsigned u = __float_as_uint(f);
  u += 0x7fffu + ((u >> 16) & 1u);   // round-to-nearest-even
  return (unsigned short)(u >> 16);
}

__device__ __forceinline__ void atomAddF(float* p, float v) {
  unsafeAtomicAdd(p, v);             // native global_atomic_add_f32
}

// Monotone float <-> uint mapping so unsigned atomicMax implements float max.
__device__ __forceinline__ unsigned fkey(float f) {
  unsigned u = __float_as_uint(f);
  return (u & 0x80000000u) ? ~u : (u | 0x80000000u);
}
__device__ __forceinline__ float fdec(unsigned k) {
  return (k & 0x80000000u) ? __uint_as_float(k & 0x7fffffffu) : __uint_as_float(~k);
}

// ---------------------------------------------------------------- init ------
__global__ void __launch_bounds__(256) init_kernel(float* __restrict__ agg,
                                                   unsigned* __restrict__ mkey,
                                                   float* __restrict__ sbuf) {
  const int idx = blockIdx.x * 256 + threadIdx.x;
  if (idx < B_ * N_ * HID_) agg[idx] = 0.f;
  if (idx < B_ * N_ * H_) { mkey[idx] = 0u; sbuf[idx] = 0.f; }
}

// ------------------------------------------------ x f32 -> bf16 convert -----
// 4 elements per thread; 1.28M threads cover B*N*D = 5.12M values.
__global__ void __launch_bounds__(256) conv_bf16_kernel(
    const float* __restrict__ in, unsigned short* __restrict__ outp) {
  const long idx = (long)blockIdx.x * 256 + threadIdx.x;
  const float4 v = *(const float4*)(in + idx * 4);
  ushort4 o;
  o.x = f2bf(v.x); o.y = f2bf(v.y); o.z = f2bf(v.z); o.w = f2bf(v.w);
  *(ushort4*)(outp + idx * 4) = o;
}

// ------------------------------------------------- weight bf16 swizzle ------
// Per 128x128 matrix: 32 tiles (kt 0..3, nt 0..7); tile = WMMA B fragment,
// lane-major, lane l holds 16 halves s=0..15:
//   l<16 : n = nt*16+l,     k = kt*32 + (s<8 ? s : s + 8)
//   l>=16: n = nt*16+l-16,  k = kt*32 + (s<8 ? s + 8 : s + 16)
__global__ void __launch_bounds__(256) swizzle_w_kernel(
    const float* __restrict__ Wq, const float* __restrict__ Wk,
    const float* __restrict__ Wv, const float* __restrict__ Wo,
    unsigned short* __restrict__ outw) {
  const int idx  = blockIdx.x * 256 + threadIdx.x;   // 0..65535
  const int mat  = idx >> 14;
  const int rlo  = idx & 16383;
  const int tile = rlo >> 9;
  const int lane = (rlo >> 4) & 31;
  const int s    = rlo & 15;
  const int kt = tile >> 3, nt = tile & 7;
  const int k = kt * 32 + ((s < 8) ? s : s + 8) + ((lane < 16) ? 0 : 8);
  const int n = nt * 16 + (lane & 15);
  const float* W = (mat == 0) ? Wq : (mat == 1) ? Wk : (mat == 2) ? Wv : Wo;
  outw[idx] = f2bf(W[k * 128 + n]);
}

// ------------------------------------------------------------ QKV GEMM ------
// 2500 blocks x 256 threads; block = one 16-row strip (B*N = 40000 rows).
// A-tile (16x32 bf16) staged into LDS by the Tensor Data Mover (wave 0 issues
// tensor_load_to_lds; TENSORcnt + workgroup barrier for visibility).
// Wave w computes cols [16w,16w+16) of Q, K and V (3 accumulators, 12 WMMAs).
__global__ void __launch_bounds__(256) qkv_gemm_kernel(
    const unsigned short* __restrict__ xb, const unsigned short* __restrict__ wswz,
    const float* __restrict__ bq, const float* __restrict__ bk,
    const float* __restrict__ bv,
    float* __restrict__ Q, float* __restrict__ K, float* __restrict__ V) {
  __shared__ unsigned short As[16][32];
  const int tid  = threadIdx.x;
  const int wave = tid >> 5, lane = tid & 31;
  const int half = lane >> 4, r = lane & 15;
  const long r0  = (long)blockIdx.x * 16;
#if HAVE_TDM
  const unsigned ldsAs = (unsigned)(uintptr_t)&As[0][0];
#endif

  v8f cq = {}, ck = {}, cv = {};
  for (int kt = 0; kt < 4; ++kt) {
    __syncthreads();                        // previous-iteration reads done
#if HAVE_TDM
    if (wave == 0) {
      const unsigned long long ga =
          (unsigned long long)(uintptr_t)xb + ((unsigned long long)(r0 * 128 + kt * 32)) * 2ull;
      v4u g0;
      g0[0] = 1u;                                            // count=1, user mode
      g0[1] = ldsAs;                                         // lds_addr
      g0[2] = (unsigned)(ga & 0xffffffffu);                  // global_addr[31:0]
      g0[3] = (unsigned)((ga >> 32) & 0x1ffffffu) | (2u << 30); // addr[56:32] | type=2
      v8i_t g1;
      g1[0] = (int)(1u << 16);       // workgroup_mask=0, data_size=1 (2 bytes)
      g1[1] = (int)(128u << 16);     // tensor_dim0[15:0] = 128 (elems/row)
      g1[2] = (int)(40000u << 16);   // tensor_dim0[31:16]=0 | tensor_dim1[15:0]=40000
      g1[3] = (int)(32u << 16);      // tensor_dim1[31:16]=0 | tile_dim0=32
      g1[4] = 16;                    // tile_dim1=16, tile_dim2=0
      g1[5] = 128;                   // tensor_dim0_stride[31:0] = 128
      g1[6] = 0;                     // stride0 hi / dim1_stride lo
      g1[7] = 0;
      v4i_t z4 = {0, 0, 0, 0};
      v8i_t z8 = {0, 0, 0, 0, 0, 0, 0, 0};
      __builtin_amdgcn_tensor_load_to_lds(g0, g1, z4, z4, z8, 0);
      __builtin_amdgcn_s_wait_tensorcnt(0);
    }
#else
#pragma unroll
    for (int t = 0; t < 2; ++t) {
      const int idx = tid + t * 256;
      const int row = idx >> 5, col = idx & 31;
      As[row][col] = xb[(r0 + row) * 128 + kt * 32 + col];
    }
#endif
    __syncthreads();                        // tile visible to all waves
    Frag16 a;
    a.q[0] = *(const uint4*)&As[r][half * 8];
    a.q[1] = *(const uint4*)&As[r][16 + half * 8];
    const unsigned short* bp = wswz + (kt * 8 + wave) * 512 + lane * 16;
    Frag16 b0, b1, b2;
    b0.q[0] = *(const uint4*)(bp);          b0.q[1] = *(const uint4*)(bp + 8);
    b1.q[0] = *(const uint4*)(bp + 16384);  b1.q[1] = *(const uint4*)(bp + 16392);
    b2.q[0] = *(const uint4*)(bp + 32768);  b2.q[1] = *(const uint4*)(bp + 32776);
    cq = __builtin_amdgcn_wmma_f32_16x16x32_bf16(false, a.v, false, b0.v, (short)0, cq, false, false);
    ck = __builtin_amdgcn_wmma_f32_16x16x32_bf16(false, a.v, false, b1.v, (short)0, ck, false, false);
    cv = __builtin_amdgcn_wmma_f32_16x16x32_bf16(false, a.v, false, b2.v, (short)0, cv, false, false);
  }
  const int col = wave * 16 + r;
  const float biq = bq[col], bik = bk[col], biv = bv[col];
#pragma unroll
  for (int i = 0; i < 8; ++i) {
    const long row = r0 + half * 8 + i;
    Q[row * 128 + col] = cq[i] + biq;
    K[row * 128 + col] = ck[i] + bik;
    V[row * 128 + col] = cv[i] + biv;
  }
}

// ---------------------------------------------------- pass E1: logits -------
// One wave per edge. Inline FiLM GEMM (edge_attr @ Wf + bf) into LDS, then
// per-batch FiLM'd q.k dot per head (8-lane groups) + segment atomic max.
__global__ void __launch_bounds__(256) edge_logits_kernel(
    const float* __restrict__ Q, const float* __restrict__ K,
    const int* __restrict__ eidx, const float* __restrict__ ea,
    const float* __restrict__ Wf, const float* __restrict__ bfv,
    float* __restrict__ logits, unsigned* __restrict__ mkey) {
  __shared__ float film[8][256];
  const int wave = threadIdx.x >> 5, lane = threadIdx.x & 31;
  const int e = blockIdx.x * 8 + wave;

  const int o0 = lane * 8;
  float acc[8];
#pragma unroll
  for (int j = 0; j < 8; ++j) acc[j] = bfv[o0 + j];
  const float* eap = ea + (long)e * 16;
#pragma unroll
  for (int i = 0; i < 16; ++i) {
    const float  ai = eap[i];
    const float4 w0 = *(const float4*)(Wf + i * 256 + o0);
    const float4 w1 = *(const float4*)(Wf + i * 256 + o0 + 4);
    acc[0] += ai * w0.x; acc[1] += ai * w0.y; acc[2] += ai * w0.z; acc[3] += ai * w0.w;
    acc[4] += ai * w1.x; acc[5] += ai * w1.y; acc[6] += ai * w1.z; acc[7] += ai * w1.w;
  }
#pragma unroll
  for (int j = 0; j < 8; ++j) film[wave][o0 + j] = acc[j];
  __syncthreads();

  const int src = eidx[e], dst = eidx[E_ + e];
  const int d0 = lane * 4;
  const float4 gm = *(const float4*)&film[wave][d0];
  const float4 bt = *(const float4*)&film[wave][128 + d0];
#pragma unroll
  for (int b = 0; b < B_; ++b) {
    const float4 q = *(const float4*)(Q + ((long)b * N_ + dst) * 128 + d0);
    const float4 k = *(const float4*)(K + ((long)b * N_ + src) * 128 + d0);
    float p = q.x * (k.x * (1.f + gm.x) + bt.x)
            + q.y * (k.y * (1.f + gm.y) + bt.y)
            + q.z * (k.z * (1.f + gm.z) + bt.z)
            + q.w * (k.w * (1.f + gm.w) + bt.w);
    p += __shfl_xor(p, 1);
    p += __shfl_xor(p, 2);
    p += __shfl_xor(p, 4);
    if ((lane & 7) == 0) {
      const int h = lane >> 3;
      const float lg = p * 0.17677669529663687f;   // 1/sqrt(32)
      logits[((long)b * E_ + e) * 4 + h] = lg;
      atomicMax(&mkey[((long)b * N_ + dst) * 4 + h], fkey(lg));
    }
  }
}

// ---------------------------------------------- pass E2: exp + seg-sum ------
__global__ void __launch_bounds__(256) edge_softmax_kernel(
    const int* __restrict__ eidx, float* __restrict__ logits,
    const unsigned* __restrict__ mkey, float* __restrict__ sbuf) {
  const int idx = blockIdx.x * 256 + threadIdx.x;  // < B*E*H
  const int h  = idx & 3;
  const int be = idx >> 2;
  const int b  = be / E_;
  const int e  = be - b * E_;
  const int dst = eidx[E_ + e];
  const float lg = logits[idx];
  const float mm = fdec(mkey[((long)b * N_ + dst) * 4 + h]);
  const float ex = __expf(lg - mm);
  logits[idx] = ex;
  atomAddF(&sbuf[((long)b * N_ + dst) * 4 + h], ex);
}

// ---------------------------------------- pass E3: normalize + scatter ------
__global__ void __launch_bounds__(256) edge_agg_kernel(
    const float* __restrict__ V, const int* __restrict__ eidx,
    const float* __restrict__ exs, const float* __restrict__ sbuf,
    float* __restrict__ agg) {
  const int wave = threadIdx.x >> 5, lane = threadIdx.x & 31;
  const int e = blockIdx.x * 8 + wave;
  const int src = eidx[e], dst = eidx[E_ + e];
  const int d0 = lane * 4, h = lane >> 3;
#pragma unroll
  for (int b = 0; b < B_; ++b) {
    const float ex  = exs[((long)b * E_ + e) * 4 + h];
    const float s   = sbuf[((long)b * N_ + dst) * 4 + h];
    const float att = ex / s;
    const float4 v = *(const float4*)(V + ((long)b * N_ + src) * 128 + d0);
    float* ap = agg + ((long)b * N_ + dst) * 128 + d0;
    atomAddF(ap + 0, att * v.x);
    atomAddF(ap + 1, att * v.y);
    atomAddF(ap + 2, att * v.z);
    atomAddF(ap + 3, att * v.w);
  }
}

// ----------------------------------- output proj + residual + LayerNorm ----
__global__ void __launch_bounds__(256) out_ln_kernel(
    const float* __restrict__ agg, const unsigned short* __restrict__ wo,
    const float* __restrict__ bo, const float* __restrict__ x,
    const float* __restrict__ lng, const float* __restrict__ lnb,
    float* __restrict__ out) {
  __shared__ unsigned short As[16][32];
  __shared__ float hb[16][128];
  const int tid  = threadIdx.x;
  const int wave = tid >> 5, lane = tid & 31;
  const int half = lane >> 4, r = lane & 15;
  const long r0  = (long)blockIdx.x * 16;

  v8f c = {};
  for (int kt = 0; kt < 4; ++kt) {
    __syncthreads();
#pragma unroll
    for (int t = 0; t < 2; ++t) {
      const int idx = tid + t * 256;
      const int row = idx >> 5, col = idx & 31;
      As[row][col] = f2bf(agg[(r0 + row) * 128 + kt * 32 + col]);
    }
    __syncthreads();
    Frag16 a, b;
    a.q[0] = *(const uint4*)&As[r][half * 8];
    a.q[1] = *(const uint4*)&As[r][16 + half * 8];
    const unsigned short* bp = wo + (kt * 8 + wave) * 512 + lane * 16;
    b.q[0] = *(const uint4*)(bp);
    b.q[1] = *(const uint4*)(bp + 8);
    c = __builtin_amdgcn_wmma_f32_16x16x32_bf16(false, a.v, false, b.v, (short)0, c, false, false);
  }
  const int col = wave * 16 + r;
  const float bval = bo[col];
#pragma unroll
  for (int i = 0; i < 8; ++i) hb[half * 8 + i][col] = c[i] + bval;
  __syncthreads();

#pragma unroll
  for (int rr = 0; rr < 2; ++rr) {
    const int m = wave * 2 + rr;
    const int j = lane * 4;
    float4 hv = *(const float4*)&hb[m][j];
    const float4 xv = *(const float4*)(x + (r0 + m) * 128 + j);
    hv.x += xv.x; hv.y += xv.y; hv.z += xv.z; hv.w += xv.w;
    float sum = hv.x + hv.y + hv.z + hv.w;
    float sq  = hv.x * hv.x + hv.y * hv.y + hv.z * hv.z + hv.w * hv.w;
#pragma unroll
    for (int o = 1; o < 32; o <<= 1) {
      sum += __shfl_xor(sum, o);
      sq  += __shfl_xor(sq, o);
    }
    const float mean = sum * (1.f / 128.f);
    const float var  = sq * (1.f / 128.f) - mean * mean;
    const float inv  = rsqrtf(var + 1e-5f);
    const float4 gv  = *(const float4*)(lng + j);
    const float4 bv2 = *(const float4*)(lnb + j);
    float4 ov;
    ov.x = (hv.x - mean) * inv * gv.x + bv2.x;
    ov.y = (hv.y - mean) * inv * gv.y + bv2.y;
    ov.z = (hv.z - mean) * inv * gv.z + bv2.z;
    ov.w = (hv.w - mean) * inv * gv.w + bv2.w;
    *(float4*)(out + (r0 + m) * 128 + j) = ov;
  }
}

// ---------------------------------------------------------------------------
extern "C" void kernel_launch(void* const* d_in, const int* in_sizes, int n_in,
                              void* d_out, int out_size, void* d_ws, size_t ws_size,
                              hipStream_t stream) {
  (void)in_sizes; (void)n_in; (void)out_size; (void)ws_size;
  const float* x   = (const float*)d_in[0];
  const int*   ei  = (const int*)d_in[1];
  const float* ea  = (const float*)d_in[2];
  const float* Wq  = (const float*)d_in[3];
  const float* bq  = (const float*)d_in[4];
  const float* Wk  = (const float*)d_in[5];
  const float* bk  = (const float*)d_in[6];
  const float* Wv  = (const float*)d_in[7];
  const float* bv  = (const float*)d_in[8];
  const float* Wf  = (const float*)d_in[9];
  const float* bfv = (const float*)d_in[10];
  const float* Wo  = (const float*)d_in[11];
  const float* bo  = (const float*)d_in[12];
  const float* lng = (const float*)d_in[13];
  const float* lnb = (const float*)d_in[14];
  float* out = (float*)d_out;

  char* ws = (char*)d_ws;
  float*          Qb   = (float*)(ws + 0);          // 20.48 MB
  float*          Kb   = (float*)(ws + 20480000);   // 20.48 MB
  float*          Vb   = (float*)(ws + 40960000);   // 20.48 MB
  float*          agg  = (float*)(ws + 61440000);   // 20.48 MB
  float*          exb  = (float*)(ws + 81920000);   // 10.24 MB (logits, then exp)
  unsigned*       mkey = (unsigned*)(ws + 92160000);// 0.64 MB
  float*          sbuf = (float*)(ws + 92800000);   // 0.64 MB
  unsigned short* wswz = (unsigned short*)(ws + 93440000); // 128 KB bf16 weights
  unsigned short* xb   = (unsigned short*)(ws + 93571072); // 10.24 MB x in bf16

  init_kernel<<<20000, 256, 0, stream>>>(agg, mkey, sbuf);
  conv_bf16_kernel<<<5000, 256, 0, stream>>>(x, xb);
  swizzle_w_kernel<<<256, 256, 0, stream>>>(Wq, Wk, Wv, Wo, wswz);
  qkv_gemm_kernel<<<2500, 256, 0, stream>>>(xb, wswz, bq, bk, bv, Qb, Kb, Vb);
  edge_logits_kernel<<<20000, 256, 0, stream>>>(Qb, Kb, ei, ea, Wf, bfv, exb, mkey);
  edge_softmax_kernel<<<10000, 256, 0, stream>>>(ei, exb, mkey, sbuf);
  edge_agg_kernel<<<20000, 256, 0, stream>>>(Vb, ei, exb, sbuf, agg);
  out_ln_kernel<<<2500, 256, 0, stream>>>(agg, wswz + 49152, bo, x, lng, lnb, out);
}
